// GCNNodeClassification_79980880986187
// MI455X (gfx1250) — matrix operand order repaired and checked by reading, compiled, not run on hardware
//
#include <hip/hip_runtime.h>
#include <math.h>

typedef __attribute__((ext_vector_type(16))) _Float16 v16h;
typedef __attribute__((ext_vector_type(8)))  _Float16 v8h;
typedef __attribute__((ext_vector_type(4)))  _Float16 v4h;
typedef __attribute__((ext_vector_type(8)))  float    v8f;

union AFrag { v16h v; v8h h[2]; };

// ---------------------------------------------------------------------------
// Small utility kernels
// ---------------------------------------------------------------------------
__global__ __launch_bounds__(256) void k_fill_f32(float* __restrict__ p, float v, int n) {
  int i = blockIdx.x * blockDim.x + threadIdx.x;
  if (i < n) p[i] = v;
}

__global__ __launch_bounds__(256) void k_zero_f32(float* __restrict__ p, int n) {
  int i = blockIdx.x * blockDim.x + threadIdx.x;
  if (i < n) p[i] = 0.0f;
}

// deg[dst] += 1 per edge (deg buffer pre-filled with 2.0 for improved=True self loops)
__global__ __launch_bounds__(256) void k_degree(const int* __restrict__ dst,
                                                float* __restrict__ deg, int E) {
  int e = blockIdx.x * blockDim.x + threadIdx.x;
  if (e < E) unsafeAtomicAdd(&deg[dst[e]], 1.0f);
}

__global__ __launch_bounds__(256) void k_rsqrt_inplace(float* __restrict__ p, int n) {
  int i = blockIdx.x * blockDim.x + threadIdx.x;
  if (i < n) p[i] = rsqrtf(p[i]);
}

// ---------------------------------------------------------------------------
// Pack weights W[128 x ncols] (f32) into WMMA B-fragment order (f16, padded):
//   WB[((t*4 + s)*32 + l)*16 + j] = W[s*32 + (l>>4)*16 + j][t*16 + (l&15)]
// so each lane's fragment for (tile t, k-step s) is 32 contiguous bytes.
// ---------------------------------------------------------------------------
__global__ __launch_bounds__(256) void k_pack_B(const float* __restrict__ W,
                                                _Float16* __restrict__ WB,
                                                int ncols, int ntiles) {
  int i = blockIdx.x * blockDim.x + threadIdx.x;
  int total = ntiles * 2048;            // ntiles * 4 steps * 32 lanes * 16 halves
  if (i >= total) return;
  int j = i & 15;
  int l = (i >> 4) & 31;
  int s = (i >> 9) & 3;
  int t = i >> 11;
  int k   = s * 32 + ((l >> 4) << 4) + j;
  int col = (t << 4) + (l & 15);
  WB[i] = (col < ncols) ? (_Float16)W[k * ncols + col] : (_Float16)0.0f;
}

// ---------------------------------------------------------------------------
// WMMA GEMM:  XW[nrows x ncols] = X[nrows x 128] * W[128 x ncols]
// Block = 256 threads = 8 waves; block covers 32 rows. Wave w owns the 16-col
// tile w; two accumulators (rows 0..15 / 16..31) so each B fragment feeds two
// v_wmma_f32_16x16x32_f16. A staged in LDS (2x ds_load_b128 per fragment),
// packed B streamed as 2x global_load_b128 per fragment.
// ---------------------------------------------------------------------------
__global__ __launch_bounds__(256) void k_gemm_wmma(const float* __restrict__ X,
                                                   const _Float16* __restrict__ WB,
                                                   float* __restrict__ XW,
                                                   int nrows, int ncols, int ntiles) {
  __shared__ _Float16 lsA[32 * 128];

  const int row0 = blockIdx.x * 32;

  // Stage A tile (32 x 128) as f16, vectorized: float4 in -> 4 halves out
  for (int idx4 = threadIdx.x; idx4 < 32 * 32; idx4 += 256) {
    int m  = idx4 >> 5;                 // row within tile
    int k4 = (idx4 & 31) << 2;          // first of 4 k's
    float4 v = make_float4(0.f, 0.f, 0.f, 0.f);
    if (row0 + m < nrows)
      v = *(const float4*)(X + (size_t)(row0 + m) * 128 + k4);
    v4h h;
    h[0] = (_Float16)v.x; h[1] = (_Float16)v.y;
    h[2] = (_Float16)v.z; h[3] = (_Float16)v.w;
    *(v4h*)(&lsA[m * 128 + k4]) = h;
  }
  __syncthreads();

  const int wave = threadIdx.x >> 5;
  const int lane = threadIdx.x & 31;
  if (wave >= ntiles) return;           // wave-uniform: EXEC all-ones for WMMA

  const int hi = lane >> 4;
  const int lm = lane & 15;

  // A fragment: lane(M=lm,hi) -> halves [k0+hi*16 .. +15]
  const _Float16* pa0 = &lsA[lm * 128 + hi * 16];          // rows 0..15
  const _Float16* pa1 = &lsA[(16 + lm) * 128 + hi * 16];   // rows 16..31
  // Packed B: fragment for (tile=wave, step s) at 32B/lane, contiguous per wave
  const _Float16* pb = WB + ((size_t)wave * 4) * 512 + lane * 16;

  v8f acc0 = {}, acc1 = {};
#pragma unroll
  for (int s = 0; s < 4; ++s) {
    const int k0 = s * 32;
    v16h b = *(const v16h*)(pb + s * 512);
    AFrag a0, a1;
    a0.h[0] = *(const v8h*)(pa0 + k0);
    a0.h[1] = *(const v8h*)(pa0 + k0 + 8);
    a1.h[0] = *(const v8h*)(pa1 + k0);
    a1.h[1] = *(const v8h*)(pa1 + k0 + 8);
    acc0 = __builtin_amdgcn_wmma_f32_16x16x32_f16(false, a0.v, false, b,
                                                  (short)0, acc0, false, false);
    acc1 = __builtin_amdgcn_wmma_f32_16x16x32_f16(false, a1.v, false, b,
                                                  (short)0, acc1, false, false);
  }

  // D layout: VGPR r -> M = r + 8*hi ; lane&15 -> N
  const int col = wave * 16 + lm;
  if (col < ncols) {
    // single scalar-side 64-bit address setup; rows advance by +ncols
    float* q = XW + (size_t)(row0 + (hi << 3)) * (size_t)ncols + col;
    const size_t step16 = (size_t)ncols * 16;
    if (row0 + 32 <= nrows) {           // full tile: unguarded store fast path
#pragma unroll
      for (int r = 0; r < 8; ++r) {
        q[0]      = acc0[r];
        q[step16] = acc1[r];
        q += ncols;
      }
    } else {                            // partial tail tile
#pragma unroll
      for (int r = 0; r < 8; ++r) {
        int m0 = row0 + r + (hi << 3);
        if (m0 < nrows)      q[0]      = acc0[r];
        if (m0 + 16 < nrows) q[step16] = acc1[r];
        q += ncols;
      }
    }
  }
}

// ---------------------------------------------------------------------------
// Edge aggregation: one wave per edge, float4 gather + hw f32 atomics scatter.
//   agg[dst, :] += dinv[src]*dinv[dst] * xw[src, :]
// xw is <= 51 MB -> L2-resident on the 192 MB L2.
// ---------------------------------------------------------------------------
__global__ __launch_bounds__(256) void k_edge_agg(const int* __restrict__ src,
                                                  const int* __restrict__ dst,
                                                  const float* __restrict__ dinv,
                                                  const float* __restrict__ xw,
                                                  float* __restrict__ agg,
                                                  int E, int hdim) {
  const int lane = threadIdx.x & 31;
  const int wave = (blockIdx.x * blockDim.x + threadIdx.x) >> 5;
  const int nwaves = (gridDim.x * blockDim.x) >> 5;

  for (int e = wave; e < E; e += nwaves) {
    int s = src[e];
    int d = dst[e];
    // prefetch next edge's source row (global_prefetch_b8)
    int e2 = e + nwaves;
    if (e2 < E) {
      int s2 = src[e2];
      __builtin_prefetch(xw + (size_t)s2 * hdim + lane * 4, 0, 0);
    }
    float coef = dinv[s] * dinv[d];
    const float* xr = xw + (size_t)s * hdim;
    float* ar = agg + (size_t)d * hdim;
    for (int q = lane * 4; q < hdim; q += 128) {     // hdim % 4 == 0 (128 or 40)
      float4 v = *(const float4*)(xr + q);
      unsafeAtomicAdd(&ar[q + 0], v.x * coef);
      unsafeAtomicAdd(&ar[q + 1], v.y * coef);
      unsafeAtomicAdd(&ar[q + 2], v.z * coef);
      unsafeAtomicAdd(&ar[q + 3], v.w * coef);
    }
  }
}

// ---------------------------------------------------------------------------
// Epilogue: out = agg + 2*dinv^2*xw + bias (+ residual) (optionally erf-GELU)
// One block per node: blockIdx.x = n, threadIdx.x = channel (no int division).
// ---------------------------------------------------------------------------
__global__ void k_combine(const float* __restrict__ agg,
                          const float* __restrict__ xw,
                          const float* __restrict__ dinv,
                          const float* __restrict__ bias,
                          const float* res, float* out,
                          int hdim, int apply_gelu) {
  const int n = blockIdx.x;
  const int c = threadIdx.x;
  if (c >= hdim) return;
  const size_t i = (size_t)n * hdim + c;
  float di = dinv[n];
  float v = agg[i] + 2.0f * di * di * xw[i] + bias[c];
  if (res) v += res[i];
  if (apply_gelu) v = 0.5f * v * (1.0f + erff(v * 0.70710678118654752f));
  out[i] = v;
}

// ---------------------------------------------------------------------------
// Host-side orchestration
// ---------------------------------------------------------------------------
extern "C" void kernel_launch(void* const* d_in, const int* in_sizes, int n_in,
                              void* d_out, int out_size, void* d_ws, size_t ws_size,
                              hipStream_t stream) {
  const float* x  = (const float*)d_in[0];
  const int*   ei = (const int*)d_in[1];
  const float* W0 = (const float*)d_in[2];  const float* b0 = (const float*)d_in[3];
  const float* W1 = (const float*)d_in[4];  const float* b1 = (const float*)d_in[5];
  const float* W2 = (const float*)d_in[6];  const float* b2 = (const float*)d_in[7];
  const float* W3 = (const float*)d_in[8];  const float* b3 = (const float*)d_in[9];

  const int H = 128;
  const int N = in_sizes[0] / H;
  const int E = in_sizes[1] / 2;
  const int C = in_sizes[9];
  const int CT = (C + 15) / 16;          // col tiles for the last layer

  const int* src = ei;
  const int* dst = ei + E;

  // ---- workspace carve-out (256B aligned) ----
  char* ws = (char*)d_ws;
  size_t off = 0;
  auto carve = [&](size_t bytes) -> void* {
    void* p = (void*)(ws + off);
    off = (off + bytes + 255) & ~(size_t)255;
    return p;
  };
  float*    dinv = (float*)carve((size_t)N * 4);
  float*    G    = (float*)carve((size_t)N * H * 4);     // activation buffer
  float*    XW   = (float*)carve((size_t)N * H * 4);     // x @ W
  float*    AGG  = (float*)carve((size_t)N * H * 4);     // edge-aggregated
  _Float16* WB0  = (_Float16*)carve((size_t)8  * 2048 * 2);  // packed B fragments
  _Float16* WB1  = (_Float16*)carve((size_t)8  * 2048 * 2);
  _Float16* WB2  = (_Float16*)carve((size_t)8  * 2048 * 2);
  _Float16* WB3  = (_Float16*)carve((size_t)CT * 2048 * 2);
  (void)ws_size; (void)n_in; (void)out_size;

  const int TB = 256;
  auto blocks = [&](long long n) { return (int)((n + TB - 1) / TB); };

  // ---- degree -> dinv (edge structure is layer-invariant) ----
  k_fill_f32<<<blocks(N), TB, 0, stream>>>(dinv, 2.0f, N);   // self-loop weight 2
  k_degree<<<blocks(E), TB, 0, stream>>>(dst, dinv, E);
  k_rsqrt_inplace<<<blocks(N), TB, 0, stream>>>(dinv, N);

  // ---- pack weights into f16 B-fragment order (once per launch) ----
  k_pack_B<<<blocks(8  * 2048), TB, 0, stream>>>(W0, WB0, H, 8);
  k_pack_B<<<blocks(8  * 2048), TB, 0, stream>>>(W1, WB1, H, 8);
  k_pack_B<<<blocks(8  * 2048), TB, 0, stream>>>(W2, WB2, H, 8);
  k_pack_B<<<blocks(CT * 2048), TB, 0, stream>>>(W3, WB3, C, CT);

  const int gemm_grid = (N + 31) / 32;
  const int agg_grid  = 4096;            // grid-stride waves over edges

  auto run_layer = [&](const float* in, const _Float16* WB, const float* bias,
                       const float* res, float* out, int hout, int ntiles,
                       int apply_gelu) {
    k_gemm_wmma<<<gemm_grid, TB, 0, stream>>>(in, WB, XW, N, hout, ntiles);
    k_zero_f32<<<blocks((long long)N * hout), TB, 0, stream>>>(AGG, N * hout);
    k_edge_agg<<<agg_grid, TB, 0, stream>>>(src, dst, dinv, XW, AGG, E, hout);
    k_combine<<<N, hout, 0, stream>>>(AGG, XW, dinv, bias, res, out, hout,
                                      apply_gelu);
  };

  // layer 0: g0 = gelu(gcn(x))
  run_layer(x, WB0, b0, nullptr, G, H, 8, 1);
  // layer 1: g1 = gelu(g0 + gcn(g0))   (elementwise in-place on G is safe)
  run_layer(G, WB1, b1, G, G, H, 8, 1);
  // layer 2: g2 = gelu(g1 + gcn(g1))
  run_layer(G, WB2, b2, G, G, H, 8, 1);
  // layer 3: out = gcn(g2)  -> [N, C] f32
  run_layer(G, WB3, b3, nullptr, (float*)d_out, C, CT, 0);
}
